// FPLayerCore_27290222198834
// MI455X (gfx1250) — compile-verified, matrix operand order checked by reading, not run on hardware
//
#include <hip/hip_runtime.h>
#include <hip/hip_bf16.h>

typedef __bf16 bf16;
typedef __attribute__((ext_vector_type(16))) __bf16 bf16x16;
typedef __attribute__((ext_vector_type(8)))  float  f32x8;
typedef __attribute__((ext_vector_type(4)))  int    v4i;

#define BN_EPS 1e-3f
#define LDT 40   // LDS tile row stride in bf16 elements (80B: 16B-aligned, spreads banks)

// ---- async global->LDS staging (CDNA5), with portable fallback --------------
#if __has_builtin(__builtin_amdgcn_global_load_async_to_lds_b128)
#define HAVE_ASYNC_LDS 1
#else
#define HAVE_ASYNC_LDS 0
#endif

__device__ __forceinline__ void cp16_g2l(bf16* lds_dst, const bf16* gbl_src)
{
#if HAVE_ASYNC_LDS
    __builtin_amdgcn_global_load_async_to_lds_b128(
        (__attribute__((address_space(1))) v4i*)(void*)gbl_src,
        (__attribute__((address_space(3))) v4i*)(void*)lds_dst, 0, 0);
#else
    *reinterpret_cast<float4*>(lds_dst) =
        *reinterpret_cast<const float4*>(gbl_src);
#endif
}

__device__ __forceinline__ void wait_async_all()
{
#if HAVE_ASYNC_LDS
#if __has_builtin(__builtin_amdgcn_s_wait_asynccnt)
    __builtin_amdgcn_s_wait_asynccnt(0);
#else
    asm volatile("s_wait_asynccnt 0" ::: "memory");
#endif
#endif
}

// ---------------------------------------------------------------------------
// 1) three_nn: for each (b,n) find 3 nearest of M ref points, emit idx + weights
// ---------------------------------------------------------------------------
__global__ __launch_bounds__(256)
void three_nn_kernel(const float* __restrict__ xyz1, const float* __restrict__ xyz2,
                     int* __restrict__ idx, float* __restrict__ wgt, int N, int M)
{
    __shared__ float sx[256], sy[256], sz[256];
    const int b = blockIdx.y;
    const int n = blockIdx.x * 256 + threadIdx.x;
    const float* p1 = xyz1 + ((size_t)b * N + n) * 3;
    const float qx = p1[0], qy = p1[1], qz = p1[2];
    float d0 = 3.4e38f, d1 = 3.4e38f, d2 = 3.4e38f;
    int   i0 = 0, i1 = 0, i2 = 0;
    for (int m0 = 0; m0 < M; m0 += 256) {
        const float* p2 = xyz2 + ((size_t)b * M + m0 + threadIdx.x) * 3;
        sx[threadIdx.x] = p2[0]; sy[threadIdx.x] = p2[1]; sz[threadIdx.x] = p2[2];
        __syncthreads();
        #pragma unroll 4
        for (int j = 0; j < 256; ++j) {
            const float dx = qx - sx[j], dy = qy - sy[j], dz = qz - sz[j];
            const float d = dx * dx + dy * dy + dz * dz;
            const int mj = m0 + j;
            if (d < d0)      { d2 = d1; i2 = i1; d1 = d0; i1 = i0; d0 = d; i0 = mj; }
            else if (d < d1) { d2 = d1; i2 = i1; d1 = d;  i1 = mj; }
            else if (d < d2) { d2 = d;  i2 = mj; }
        }
        __syncthreads();
    }
    d0 = fmaxf(d0, 1e-10f); d1 = fmaxf(d1, 1e-10f); d2 = fmaxf(d2, 1e-10f);
    const float w0 = 1.f / d0, w1 = 1.f / d1, w2 = 1.f / d2;
    const float s = 1.f / (w0 + w1 + w2);
    const size_t o = ((size_t)b * N + n) * 3;
    idx[o] = i0; idx[o + 1] = i1; idx[o + 2] = i2;
    wgt[o] = w0 * s; wgt[o + 1] = w1 * s; wgt[o + 2] = w2 * s;
}

// ---------------------------------------------------------------------------
// 2) weighted gather of points2 + concat points1 -> X0 bf16 [P][C2+C1]
// ---------------------------------------------------------------------------
__global__ __launch_bounds__(256)
void interp_concat_kernel(const float* __restrict__ points2, const float* __restrict__ points1,
                          const int* __restrict__ idx, const float* __restrict__ wgt,
                          bf16* __restrict__ X0, int N, int M, int C1, int C2)
{
    const int p = blockIdx.x;            // 0..B*N-1
    const int b = p / N;
    const size_t o3 = (size_t)p * 3;
    const int i0 = idx[o3], i1 = idx[o3 + 1], i2 = idx[o3 + 2];
    const float w0 = wgt[o3], w1 = wgt[o3 + 1], w2 = wgt[o3 + 2];
    const float* r0 = points2 + ((size_t)b * M + i0) * C2;
    const float* r1 = points2 + ((size_t)b * M + i1) * C2;
    const float* r2 = points2 + ((size_t)b * M + i2) * C2;
    const int c = threadIdx.x;
    bf16* xr = X0 + (size_t)p * (C2 + C1);
    const float v = w0 * r0[c] + w1 * r1[c] + w2 * r2[c];
    xr[c] = (bf16)v;
    if (c < C1) xr[C2 + c] = (bf16)points1[(size_t)p * C1 + c];
}

// ---------------------------------------------------------------------------
// 3) convert weights f32 [K][Ncol] -> transposed bf16 [Ncol][K]
// ---------------------------------------------------------------------------
__global__ __launch_bounds__(256)
void wcvt_kernel(const float* __restrict__ W, bf16* __restrict__ Wt, int K, int Ncol)
{
    const int t = blockIdx.x * 256 + threadIdx.x;
    if (t >= K * Ncol) return;
    const int k = t % K, n = t / K;
    Wt[(size_t)n * K + k] = (bf16)W[(size_t)k * Ncol + n];
}

// ---------------------------------------------------------------------------
// 4) bf16 WMMA GEMM: Y[r][n] = sum_k A[r][k] * Bt[n][k] + bias[n]
//    block tile 128 rows x 256 cols (full width), 8 waves (2m x 4n),
//    wave tile 64x64 (16 WMMA/iter), BK=32, double-buffered async LDS staging
// ---------------------------------------------------------------------------
__global__ __launch_bounds__(256)
void gemm_bf16_kernel(const bf16* __restrict__ A, const bf16* __restrict__ Bt,
                      const float* __restrict__ bias, float* __restrict__ Y,
                      int K, int NCOL)
{
    __shared__ bf16 ldsA[2][128 * LDT];
    __shared__ bf16 ldsB[2][256 * LDT];
    const int tid  = threadIdx.x;
    const int lane = tid & 31;
    const int wave = tid >> 5;        // 0..7
    const int wm   = wave >> 2;       // 0..1 : 64-row strip
    const int wn   = wave & 3;        // 0..3 : 64-col strip
    const int half = lane >> 4;       // hi/lo 16 lanes
    const int l16  = lane & 15;
    const int rowBase = blockIdx.x * 128;

    f32x8 acc[4][4];
    const f32x8 zf = {0.f, 0.f, 0.f, 0.f, 0.f, 0.f, 0.f, 0.f};
    #pragma unroll
    for (int i = 0; i < 4; ++i)
        #pragma unroll
        for (int j = 0; j < 4; ++j) acc[i][j] = zf;

    union FragU { bf16x16 v; unsigned u[8]; };

    // stage one 32-deep K-chunk (A: 128x32, B: 256x32) into buffer bi
    auto stage = [&](int kk, int bi) {
        #pragma unroll
        for (int s = 0; s < 2; ++s) {                 // A: 512 x 16B chunks
            const int chunk = tid + s * 256;
            const int r  = chunk >> 2;                // 0..127
            const int cc = (chunk & 3) * 8;           // 0,8,16,24
            cp16_g2l(&ldsA[bi][r * LDT + cc],
                     A + (size_t)(rowBase + r) * K + kk + cc);
        }
        #pragma unroll
        for (int s = 0; s < 4; ++s) {                 // B: 1024 x 16B chunks
            const int chunk = tid + s * 256;
            const int r  = chunk >> 2;                // 0..255 (= column)
            const int cc = (chunk & 3) * 8;
            cp16_g2l(&ldsB[bi][r * LDT + cc],
                     Bt + (size_t)r * K + kk + cc);
        }
    };

    stage(0, 0);
    int buf = 0;
    for (int k0 = 0; k0 < K; k0 += 32) {
        wait_async_all();          // our staging landed in LDS
        __syncthreads();           // everyone's staging landed; prev reads done
        if (k0 + 32 < K) stage(k0 + 32, buf ^ 1);

        // per-lane fragment gather, per ISA 16-bit A-matrix layout:
        // lane<16: K pairs {0..7,16..23}; lane>=16: +8
        FragU af[4], bfg[4];
        #pragma unroll
        for (int mt = 0; mt < 4; ++mt) {
            const int r = wm * 64 + mt * 16 + l16;
            #pragma unroll
            for (int i = 0; i < 8; ++i) {
                const int kb = (i & 3) * 2 + (i >> 2) * 16 + half * 8;
                af[mt].u[i] = *reinterpret_cast<const unsigned*>(&ldsA[buf][r * LDT + kb]);
            }
        }
        #pragma unroll
        for (int nt = 0; nt < 4; ++nt) {
            const int c = wn * 64 + nt * 16 + l16;
            #pragma unroll
            for (int i = 0; i < 8; ++i) {
                const int kb = (i & 3) * 2 + (i >> 2) * 16 + half * 8;
                bfg[nt].u[i] = *reinterpret_cast<const unsigned*>(&ldsB[buf][c * LDT + kb]);
            }
        }
        #pragma unroll
        for (int mt = 0; mt < 4; ++mt)
            #pragma unroll
            for (int nt = 0; nt < 4; ++nt)
                acc[mt][nt] = __builtin_amdgcn_wmma_f32_16x16x32_bf16(
                    false, af[mt].v, false, bfg[nt].v, (short)0, acc[mt][nt], false, false);
        buf ^= 1;
    }

    // epilogue: C layout — VGPR g: lanes0-15 row=g col=lane; lanes16-31 row=8+g
    #pragma unroll
    for (int mt = 0; mt < 4; ++mt)
        #pragma unroll
        for (int nt = 0; nt < 4; ++nt) {
            const int col = wn * 64 + nt * 16 + l16;
            const float bv = bias[col];
            #pragma unroll
            for (int g = 0; g < 8; ++g) {
                const int r = rowBase + wm * 64 + mt * 16 + half * 8 + g;
                Y[(size_t)r * NCOL + col] = acc[mt][nt][g] + bv;
            }
        }
}

// ---------------------------------------------------------------------------
// 5) per-column sum / sumsq partial reduction (atomics into zeroed buffers)
// ---------------------------------------------------------------------------
__global__ __launch_bounds__(256)
void colstats_kernel(const float* __restrict__ Y, float* __restrict__ sum,
                     float* __restrict__ sqs, int NCOL)
{
    const int col = threadIdx.x;
    const int r0 = blockIdx.x * 256;
    float s = 0.f, q = 0.f;
    #pragma unroll 4
    for (int r = 0; r < 256; ++r) {
        const float v = Y[(size_t)(r0 + r) * NCOL + col];
        s += v; q += v * v;
    }
    atomicAdd(&sum[col], s);
    atomicAdd(&sqs[col], q);
}

__global__ void bnparam_kernel(const float* __restrict__ sum, const float* __restrict__ sqs,
                               const float* __restrict__ gamma, const float* __restrict__ beta,
                               float* __restrict__ a, float* __restrict__ c, float inv_n)
{
    const int i = threadIdx.x;
    const float mean = sum[i] * inv_n;
    const float var  = sqs[i] * inv_n - mean * mean;
    const float s = gamma[i] * rsqrtf(var + BN_EPS);
    a[i] = s;
    c[i] = beta[i] - mean * s;
}

__global__ __launch_bounds__(256)
void bnrelu_bf16_kernel(const float* __restrict__ Y, const float* __restrict__ a,
                        const float* __restrict__ c, bf16* __restrict__ X, int NCOL)
{
    const size_t t = (size_t)blockIdx.x * 256 + threadIdx.x;
    const int col = (int)(t & (size_t)(NCOL - 1));
    X[t] = (bf16)fmaxf(a[col] * Y[t] + c[col], 0.f);
}

__global__ __launch_bounds__(256)
void bnrelu_f32_kernel(const float* __restrict__ Y, const float* __restrict__ a,
                       const float* __restrict__ c, float* __restrict__ out, int NCOL)
{
    const size_t t = (size_t)blockIdx.x * 256 + threadIdx.x;
    const int col = (int)(t & (size_t)(NCOL - 1));
    out[t] = fmaxf(a[col] * Y[t] + c[col], 0.f);
}

__global__ void zero_kernel(float* __restrict__ p, int n)
{
    const int t = blockIdx.x * 256 + threadIdx.x;
    if (t < n) p[t] = 0.f;
}

// ---------------------------------------------------------------------------
extern "C" void kernel_launch(void* const* d_in, const int* in_sizes, int n_in,
                              void* d_out, int out_size, void* d_ws, size_t ws_size,
                              hipStream_t stream)
{
    (void)in_sizes; (void)n_in; (void)out_size; (void)ws_size;
    const float* xyz1    = (const float*)d_in[0];
    const float* xyz2    = (const float*)d_in[1];
    const float* points1 = (const float*)d_in[2];
    const float* points2 = (const float*)d_in[3];
    const float* W0  = (const float*)d_in[4];
    const float* b0  = (const float*)d_in[5];
    const float* g0  = (const float*)d_in[6];
    const float* be0 = (const float*)d_in[7];
    const float* W1  = (const float*)d_in[8];
    const float* b1  = (const float*)d_in[9];
    const float* g1  = (const float*)d_in[10];
    const float* be1 = (const float*)d_in[11];

    const int B = 16, N = 4096, M = 1024, C1 = 128, C2 = 256;
    const int CIN = C2 + C1;          // 384
    const int C = 256;                // mlp width
    const int P = B * N;              // 65536 rows

    char* ws = (char*)d_ws;
    size_t off = 0;
    auto take = [&](size_t bytes) -> void* {
        void* p = ws + off;
        off = (off + bytes + 255) & ~(size_t)255;
        return p;
    };
    int*   idx  = (int*)  take((size_t)P * 3 * sizeof(int));
    float* wgt  = (float*)take((size_t)P * 3 * sizeof(float));
    bf16*  X0   = (bf16*) take((size_t)P * CIN * sizeof(bf16));
    bf16*  X1   = (bf16*) take((size_t)P * C * sizeof(bf16));
    float* Yb   = (float*)take((size_t)P * C * sizeof(float));   // reused both layers
    bf16*  W0t  = (bf16*) take((size_t)C * CIN * sizeof(bf16));
    bf16*  W1t  = (bf16*) take((size_t)C * C * sizeof(bf16));
    float* ssum = (float*)take(256 * sizeof(float));
    float* ssq  = (float*)take(256 * sizeof(float));
    float* sa   = (float*)take(256 * sizeof(float));
    float* sc   = (float*)take(256 * sizeof(float));

    const float invP = 1.f / (float)P;

    three_nn_kernel<<<dim3(N / 256, B), 256, 0, stream>>>(xyz1, xyz2, idx, wgt, N, M);
    interp_concat_kernel<<<P, 256, 0, stream>>>(points2, points1, idx, wgt, X0, N, M, C1, C2);
    wcvt_kernel<<<(CIN * C + 255) / 256, 256, 0, stream>>>(W0, W0t, CIN, C);
    wcvt_kernel<<<(C * C + 255) / 256, 256, 0, stream>>>(W1, W1t, C, C);

    // ---- layer 0: Y = X0 @ W0 + b0 ; BN ; ReLU -> X1 (bf16) ----
    zero_kernel<<<1, 256, 0, stream>>>(ssum, 256);
    zero_kernel<<<1, 256, 0, stream>>>(ssq, 256);
    gemm_bf16_kernel<<<P / 128, 256, 0, stream>>>(X0, W0t, b0, Yb, CIN, C);
    colstats_kernel<<<P / 256, 256, 0, stream>>>(Yb, ssum, ssq, C);
    bnparam_kernel<<<1, 256, 0, stream>>>(ssum, ssq, g0, be0, sa, sc, invP);
    bnrelu_bf16_kernel<<<(P * C) / 256, 256, 0, stream>>>(Yb, sa, sc, X1, C);

    // ---- layer 1: Y = X1 @ W1 + b1 ; BN ; ReLU -> out (f32) ----
    zero_kernel<<<1, 256, 0, stream>>>(ssum, 256);
    zero_kernel<<<1, 256, 0, stream>>>(ssq, 256);
    gemm_bf16_kernel<<<P / 128, 256, 0, stream>>>(X1, W1t, b1, Yb, C, C);
    colstats_kernel<<<P / 256, 256, 0, stream>>>(Yb, ssum, ssq, C);
    bnparam_kernel<<<1, 256, 0, stream>>>(ssum, ssq, g1, be1, sa, sc, invP);
    bnrelu_f32_kernel<<<(P * C) / 256, 256, 0, stream>>>(Yb, sa, sc, (float*)d_out, C);
}
